// StochasticStateModel_55250459295832
// MI455X (gfx1250) — matrix operand order repaired and checked by reading, not compiled, not run on hardware
//
#include <hip/hip_runtime.h>
#include <hip/hip_bf16.h>

// ---------------------------------------------------------------------------
// StochasticStateModel on MI455X (gfx1250, wave32, WMMA + TDM)
//
//   F = 68 features, P = 512*1024 pixels, 7 experts (68x68 each)
//   out[:,p] = W[eta'(p)] @ feat[:,p] + b[eta'(p)]
//
// 16 pixels = WMMA N dim, 5 M-tiles (68->80), 3 K-tiles of 32 (68->96),
// v_wmma_f32_16x16x32_f16. Disjoint per-column expert masks let all experts
// present in a tile accumulate into one C. A-fragments are pre-swizzled to
// workspace, DMA'd to LDS once per block via TENSOR_LOAD_TO_LDS (TDM), and
// read via ds_load_b128. Each wave processes 16 consecutive tiles and
// prefetches the next tile's feature rows during compute.
// ---------------------------------------------------------------------------

typedef _Float16 v16h  __attribute__((ext_vector_type(16)));
typedef float    v8f   __attribute__((ext_vector_type(8)));
typedef unsigned int u32x4 __attribute__((ext_vector_type(4)));
typedef int      i32x4 __attribute__((ext_vector_type(4)));
typedef int      i32x8 __attribute__((ext_vector_type(8)));

#define FDIM   68
#define NE     7
#define MT     5                      // M tiles: 80 rows padded
#define KT     3                      // K tiles: 96 cols padded
#define PPIX   (512 * 1024)
#define NTILE  (PPIX / 16)            // 32768 pixel tiles
#define FRAG_HALFS 512                // 32 lanes * 16 halfs per A fragment
#define NFRAG  (NE * MT * KT)         // 105 fragments
#define WSA_BYTES  (NFRAG * FRAG_HALFS * 2)   // 107,520 B
#define WSA_DW     (WSA_BYTES / 4)            // 26,880 dwords
#define ETA_OFF    (128 * 1024)       // new_eta int32 region in d_ws
#define TPW    16                     // tiles per wave
#define WAVES_PER_BLOCK 8

#if __has_builtin(__builtin_amdgcn_tensor_load_to_lds)
#define HAVE_TDM 1
#else
#define HAVE_TDM 0
#endif

// ------------------------- RNG (PCG hash) ----------------------------------
__device__ __forceinline__ unsigned pcg_hash(unsigned v) {
    v = v * 747796405u + 2891336453u;
    unsigned w = ((v >> ((v >> 28) + 4u)) ^ v) * 277803737u;
    return (w >> 22) ^ w;
}

// --------------------- Kernel 1: Markov eta update -------------------------
__global__ void eta_update_kernel(const int* __restrict__ eta,
                                  const float* __restrict__ T,
                                  int* __restrict__ new_eta) {
    int p = blockIdx.x * blockDim.x + threadIdx.x;
    if (p >= PPIX) return;
    int s = eta[p];
    unsigned h = pcg_hash((unsigned)p * 2654435761u ^ 42u);
    float r = (float)(h >> 8) * (1.0f / 16777216.0f);   // [0,1)
    float c = 0.0f;
    int k = NE - 1;
    #pragma unroll
    for (int j = 0; j < NE; ++j) {
        c += T[s * NE + j];
        if (r < c) { k = j; break; }
    }
    new_eta[p] = k;
}

// ---------- Kernel 2: swizzle weights into WMMA A-fragment layout ----------
// Fragment f = (e*MT + mt)*KT + kt; per ISA 7.12.2 (16-bit A 16x32):
//   lane<16 : M = mt*16+lane,    halfs 0..7 -> K=ks+0..7,  8..15 -> K=ks+16..23
//   lane>=16: M = mt*16+lane-16, halfs 0..7 -> K=ks+8..15, 8..15 -> K=ks+24..31
__global__ void weight_prep_kernel(const float* __restrict__ Wm,
                                   _Float16* __restrict__ wsA) {
    int idx  = blockIdx.x * blockDim.x + threadIdx.x;
    int frag = idx >> 5;
    int lane = idx & 31;
    if (frag >= NFRAG) return;
    int e  = frag / (MT * KT);
    int r  = frag % (MT * KT);
    int mt = r / KT;
    int kt = r % KT;

    int m  = mt * 16 + (lane & 15);
    int ks = kt * 32 + ((lane >= 16) ? 8 : 0);

    _Float16* dst = wsA + frag * FRAG_HALFS + lane * 16;
    #pragma unroll
    for (int i = 0; i < 16; ++i) {
        int k = ks + ((i < 8) ? i : (16 + (i - 8)));
        float v = 0.0f;
        if (m < FDIM && k < FDIM)
            v = Wm[(e * FDIM + m) * FDIM + k];
        dst[i] = (_Float16)v;
    }
}

// -------- TDM: DMA the whole A-fragment table global -> LDS (one D#) -------
#if HAVE_TDM
__device__ __forceinline__ void tdm_load_A_to_lds(const _Float16* gsrc,
                                                  _Float16* lds_dst) {
    const unsigned long long ga = (unsigned long long)(uintptr_t)gsrc;
    const unsigned lds_off = (unsigned)(uintptr_t)lds_dst;  // addr[31:0] = LDS offset

    u32x4 g0;
    g0.x = 0x1u;                                  // count=1 valid user descriptor
    g0.y = lds_off;                               // lds_addr (bytes)
    g0.z = (unsigned)ga;                          // global_addr[31:0]
    g0.w = (unsigned)((ga >> 32) & 0x01FFFFFFull) // global_addr[56:32]
         | 0x80000000u;                           // type=2 ("image")

    i32x8 g1;
    g1[0] = 0x00020000;                           // data_size=4B; no mcast/pad/iter
    g1[1] = (int)((WSA_DW & 0xFFFF) << 16);       // tensor_dim0[15:0]
    g1[2] = (int)(((unsigned)WSA_DW >> 16) | (1u << 16)); // dim0[31:16], dim1=1
    g1[3] = (int)((WSA_DW & 0xFFFF) << 16);       // tile_dim0 = WSA_DW
    g1[4] = 1;                                    // tile_dim1=1, tile_dim2=0
    g1[5] = (int)WSA_DW;                          // tensor_dim0_stride[31:0]
    g1[6] = (int)((WSA_DW & 0xFFFF) << 16);       // stride0 hi=0, stride1 lo
    g1[7] = 0;                                    // stride1 hi

    i32x4 z4 = {};
#if defined(__clang_major__) && (__clang_major__ >= 23)
    i32x8 z8 = {};
    __builtin_amdgcn_tensor_load_to_lds(g0, g1, z4, z4, z8, 0);
#else
    __builtin_amdgcn_tensor_load_to_lds(g0, g1, z4, z4, 0);
#endif
    __builtin_amdgcn_s_wait_tensorcnt(0);
}
#endif

// ------------------- Kernel 3: expert-masked WMMA GEMM ---------------------
__global__ void __launch_bounds__(256)
moe_wmma_kernel(const float* __restrict__ feat,      // [68][P]
                const _Float16* __restrict__ wsA,
                const int* __restrict__ new_eta,
                const float* __restrict__ bias,      // [7][68]
                float* __restrict__ out) {           // [68][P]
    __shared__ __align__(32) _Float16 sA[NFRAG * FRAG_HALFS];   // 107,520 B
    __shared__ float sBias[NE * FDIM];                          //   1,904 B

    const int tid  = threadIdx.x;
    const int wave = tid >> 5;
    const int lane = tid & 31;
    const int col  = lane & 15;
    const bool hi  = lane >= 16;

    // ---- stage A-fragment table + bias -> LDS (once per block) ----
#if HAVE_TDM
    if (wave == 0)
        tdm_load_A_to_lds(wsA, sA);       // one TDM DMA; EXEC-independent
    for (int i = tid; i < NE * FDIM; i += 256) sBias[i] = bias[i];
#else
    {
        const uint4* src = (const uint4*)wsA;
        uint4*       dst = (uint4*)sA;
        for (int i = tid; i < WSA_BYTES / 16; i += 256) dst[i] = src[i];
        for (int i = tid; i < NE * FDIM; i += 256)      sBias[i] = bias[i];
    }
#endif
    __syncthreads();

    const v16h* sFrag = (const v16h*)sA;   // 32 v16h units per fragment
    const v16h  zfrag = {};

    // each wave owns 16 consecutive tiles = 256 consecutive pixels
    const int tile0 = (blockIdx.x * WAVES_PER_BLOCK + wave) * TPW;

    for (int it = 0; it < TPW; ++it) {
        const int p0 = (tile0 + it) * 16;
        const int p  = p0 + col;

        // prefetch next tile's feature rows into near cache (3 insts/tile)
        if (it + 1 < TPW) {
            #pragma unroll
            for (int i = 0; i < 3; ++i) {
                const int k = lane + i * 32;
                if (k < FDIM)
                    __builtin_prefetch(feat + (size_t)k * PPIX + p0 + 16, 0, 0);
            }
        }

        const int col_e = new_eta[p];   // expert owning this lane's column

        // ---- build B fragments (f32 -> f16, WMMA B layout) ----
        v16h b[KT];
        #pragma unroll
        for (int kt = 0; kt < KT; ++kt) {
            const int ks = kt * 32 + (hi ? 8 : 0);
            #pragma unroll
            for (int i = 0; i < 16; ++i) {
                const int k = ks + ((i < 8) ? i : (16 + (i - 8)));
                float v = 0.0f;
                if (k < FDIM)
                    v = feat[(size_t)k * PPIX + p];   // coalesced 64B/half-wave
                b[kt][i] = (_Float16)v;
            }
        }

        v8f acc[MT];
        #pragma unroll
        for (int mt = 0; mt < MT; ++mt) acc[mt] = {};

        // ---- accumulate experts present in tile (disjoint column masks) ----
        for (int e = 0; e < NE; ++e) {
            unsigned present = __builtin_amdgcn_ballot_w32(col_e == e);
            if (present == 0u) continue;       // uniform branch: EXEC stays full

            const bool mine = (col_e == e);
            v16h be[KT];
            #pragma unroll
            for (int kt = 0; kt < KT; ++kt)
                be[kt] = mine ? b[kt] : zfrag;

            #pragma unroll
            for (int mt = 0; mt < MT; ++mt) {
                const int fbase = (e * MT + mt) * KT;
                #pragma unroll
                for (int kt = 0; kt < KT; ++kt) {
                    v16h a = sFrag[(size_t)(fbase + kt) * 32 + lane]; // ds_load
                    acc[mt] = __builtin_amdgcn_wmma_f32_16x16x32_f16(
                        /*neg_a=*/false, a, /*neg_b=*/false, be[kt],
                        /*c_mod=*/(short)0, acc[mt],
                        /*reuse_a=*/false, /*reuse_b=*/false);
                }
            }
        }

        // ---- add bias (LDS gather), store coalesced ----
        #pragma unroll
        for (int mt = 0; mt < MT; ++mt) {
            #pragma unroll
            for (int j = 0; j < 8; ++j) {
                const int m = mt * 16 + j + (hi ? 8 : 0);
                if (m < FDIM)
                    out[(size_t)m * PPIX + p] = acc[mt][j] + sBias[col_e * FDIM + m];
            }
        }
    }
}

// ---------------------------------------------------------------------------
extern "C" void kernel_launch(void* const* d_in, const int* in_sizes, int n_in,
                              void* d_out, int out_size, void* d_ws, size_t ws_size,
                              hipStream_t stream) {
    const float* x    = (const float*)d_in[0];   // [2,34,512,1024] -> [68][P]
    const float* Wm   = (const float*)d_in[1];   // [7,68,68]
    const float* b    = (const float*)d_in[2];   // [7,68]
    const float* T    = (const float*)d_in[3];   // [7,7]
    const int*   eta  = (const int*)d_in[4];     // [512,1024]
    float*       out  = (float*)d_out;           // [68][P]

    _Float16* wsA     = (_Float16*)d_ws;
    int*      new_eta = (int*)((char*)d_ws + ETA_OFF);

    // 1) per-pixel Markov transition sampling
    eta_update_kernel<<<PPIX / 256, 256, 0, stream>>>(eta, T, new_eta);

    // 2) weights -> WMMA A-fragment layout (tiny, L2-resident afterwards)
    {
        const int nthr = NFRAG * 32;             // 3360
        weight_prep_kernel<<<(nthr + 95) / 96, 96, 0, stream>>>(Wm, wsA);
    }

    // 3) masked-expert WMMA GEMM: 8 waves/block, 16 tiles/wave, TDM-staged LDS
    moe_wmma_kernel<<<NTILE / (WAVES_PER_BLOCK * TPW), 256, 0, stream>>>(
        x, wsA, new_eta, b, out);
}
